// WaveRNN_48911087567376
// MI455X (gfx1250) — matrix operand order repaired and checked by reading, not compiled, not run on hardware
//
#include <hip/hip_runtime.h>

// ---------------------------------------------------------------------------
// Types
// ---------------------------------------------------------------------------
typedef __bf16 bhalf;
typedef __attribute__((ext_vector_type(16))) __bf16 v16bf;
typedef __attribute__((ext_vector_type(8)))  __bf16 v8bf;
typedef __attribute__((ext_vector_type(8)))  float  v8f;
typedef unsigned int v4u __attribute__((ext_vector_type(4)));
typedef int v8i __attribute__((ext_vector_type(8)));
typedef int v4i __attribute__((ext_vector_type(4)));

#define HDIM   1024
#define DHALF  512
#define EMB    128
#define CONDD  512
#define LSEQ   4096
#define NBATCH 2
#define KIN    896          // COND + 3*EMB
#define N3H    3072
#define EPSBN  1e-5f
#define NROWS  (NBATCH*LSEQ)   // 8192

#define SCAN_NWG 64

// Toolchain-adaptive TDM builtin arity (therock headers => 6-arg form).
#if __has_include(<hip/amd_detail/amd_gfx1250_TDM.h>)
#define TDM_6ARG 1
#else
#define TDM_6ARG 0
#endif

static __device__ __forceinline__ v8f wmma_bf16(v16bf a, v16bf b, v8f c) {
  return __builtin_amdgcn_wmma_f32_16x16x32_bf16(false, a, false, b, (short)0, c,
                                                 false, false);
}
static __device__ __forceinline__ float sigmoidf_(float x) {
  return 1.0f / (1.0f + __expf(-x));
}

// ---------------------------------------------------------------------------
// Tensor Data Mover: async 2D tile load global->LDS (bf16 elements).
// LDS padding via descriptor: 32 DWORDs data (one 64-elem bf16 row) + 4 DWORDs
// pad -> LDS row stride of 72 bf16 elements (144B: 16B aligned, and 36-DWORD
// stride has no 64-bank conflicts for lane distances < 16).
// ---------------------------------------------------------------------------
static __device__ __forceinline__ void tdm_load_2d_padded(
    unsigned lds_addr, const void* gptr, unsigned tile_x, unsigned tile_y,
    unsigned long long stride_elems) {
  unsigned long long ga = (unsigned long long)(size_t)gptr;
  v4u g0;
  g0[0] = 1u;                                    // count=1, user descriptor
  g0[1] = lds_addr;                              // LDS byte address
  g0[2] = (unsigned)ga;                          // global_addr[31:0]
  g0[3] = (unsigned)((ga >> 32) & 0x01FFFFFFu)   // global_addr[56:32]
          | (2u << 30);                          // type = 2 ("image")
  unsigned td0 = 1u << 30, td1 = 1u << 30;       // huge tensor dims (no OOB)
  v8i g1;
  g1[0] = (int)((1u << 16)                       // data_size = 1 (2 bytes)
                | (1u << 20)                     // pad_enable
                | (4u << 22)                     // pad_interval: 32 DWORDs
                | (3u << 25));                   // pad_amount: 4 DWORDs
  g1[1] = (int)((td0 & 0xFFFFu) << 16);                           // dim0 lo16
  g1[2] = (int)(((td0 >> 16) & 0xFFFFu) | ((td1 & 0xFFFFu) << 16));
  g1[3] = (int)(((td1 >> 16) & 0xFFFFu) | (tile_x << 16));        // tile_dim0
  g1[4] = (int)(tile_y & 0xFFFFu);               // tile_dim1 (tile_dim2 = 0)
  g1[5] = (int)(unsigned)stride_elems;           // tensor_dim0_stride lo32
  g1[6] = (int)((unsigned)(stride_elems >> 32) & 0xFFFFu);
  g1[7] = 0;
  v4i z4 = {0, 0, 0, 0};
#if TDM_6ARG
  v8i z8 = {0, 0, 0, 0, 0, 0, 0, 0};
  __builtin_amdgcn_tensor_load_to_lds(g0, g1, z4, z4, z8, 0);
#else
  __builtin_amdgcn_tensor_load_to_lds(g0, g1, z4, z4, 0);
#endif
}

// ---------------------------------------------------------------------------
// Small upsample-network kernels (negligible FLOPs, VALU is fine)
// ---------------------------------------------------------------------------
__global__ void k_conv_in(const float* __restrict__ mel,
                          const float* __restrict__ w,  // [3,80,512]
                          float* __restrict__ y) {      // [2,22,512]
  int o = blockIdx.x * blockDim.x + threadIdx.x;
  if (o >= 2 * 22 * 512) return;
  int c = o & 511, l = (o >> 9) % 22, n = o / (22 * 512);
  float acc = 0.f;
  for (int kw = 0; kw < 3; kw++)
    for (int i = 0; i < 80; i++)
      acc += mel[(n * 24 + l + kw) * 80 + i] * w[(kw * 80 + i) * 512 + c];
  y[o] = acc;
}

__global__ void k_dconv(const float* __restrict__ x,   // [2,Lin,512]
                        const float* __restrict__ w,   // [2,512,512]
                        float* __restrict__ y,         // [2,Lout,512]
                        int Lin, int Lout, int rate) {
  int o = blockIdx.x * blockDim.x + threadIdx.x;
  if (o >= 2 * Lout * 512) return;
  int c = o & 511, l = (o >> 9) % Lout, n = o / (Lout * 512);
  const float* x0 = x + (long)(n * Lin + l) * 512;
  const float* x1 = x + (long)(n * Lin + l + rate) * 512;
  float acc = 0.f;
  for (int i = 0; i < 512; i++)
    acc += x0[i] * w[i * 512 + c] + x1[i] * w[(512 + i) * 512 + c];
  y[o] = acc;
}

__global__ void k_resadd(const float* __restrict__ x,  // [2,Lin,512]
                         const float* __restrict__ r,  // [2,Lout,512]
                         float* __restrict__ y,        // [2,Lout,512]
                         int Lin, int Lout, int off) {
  int o = blockIdx.x * blockDim.x + threadIdx.x;
  if (o >= 2 * Lout * 512) return;
  int c = o & 511, l = (o >> 9) % Lout, n = o / (Lout * 512);
  y[o] = x[(long)(n * Lin + l + off) * 512 + c] + r[o];
}

__global__ void k_upsample(const float* __restrict__ x,  // [2,Lin,512]
                           const float* __restrict__ w,  // [512,512]
                           float* __restrict__ y,        // [2,Lin*stride,512]
                           int Lin, int stride) {
  int Lout = Lin * stride;
  int o = blockIdx.x * blockDim.x + threadIdx.x;
  if (o >= 2 * Lout * 512) return;
  int c = o & 511, l = (o >> 9) % Lout, n = o / (Lout * 512);
  float acc = 0.f;
  if ((l % stride) == 0) {
    const float* xr = x + (long)(n * Lin + l / stride) * 512;
    for (int k = 0; k < 512; k++) acc += xr[k] * w[k * 512 + c];
  }
  y[o] = acc;
}

// BatchNorm (training stats over N,L) + ReLU, in place. One block per channel.
__global__ void k_bn_relu(float* __restrict__ xb,
                          const float* __restrict__ s,
                          const float* __restrict__ o,
                          int C, int NL) {
  __shared__ float s1[256], s2[256];
  int c = blockIdx.x, t = threadIdx.x;
  float a = 0.f, b = 0.f;
  for (int i = t; i < NL; i += 256) {
    float v = xb[(long)i * C + c];
    a += v; b += v * v;
  }
  s1[t] = a; s2[t] = b; __syncthreads();
  for (int st = 128; st > 0; st >>= 1) {
    if (t < st) { s1[t] += s1[t + st]; s2[t] += s2[t + st]; }
    __syncthreads();
  }
  float mean = s1[0] / (float)NL;
  float var  = s2[0] / (float)NL - mean * mean;
  float inv  = rsqrtf(var + EPSBN);
  float sc = s[c], of = o[c];
  for (int i = t; i < NL; i += 256) {
    long idx = (long)i * C + c;
    float v = sc * (xb[idx] - mean) * inv + of;
    xb[idx] = v > 0.f ? v : 0.f;
  }
}

// ---------------------------------------------------------------------------
// Build GRU input rows (mel_up | c_embed[coarse] | f_embed[fine] | c_embed[ct])
// as bf16 [8192, 896]
// ---------------------------------------------------------------------------
__global__ void k_build_inputs(const float* __restrict__ u3,     // [2,256,512]
                               const int* __restrict__ xi,       // [2,4096,3]
                               const float* __restrict__ cemb,   // [256,128]
                               const float* __restrict__ femb,   // [256,128]
                               bhalf* __restrict__ dst) {        // [8192,896]
  long idx = (long)blockIdx.x * blockDim.x + threadIdx.x;
  if (idx >= (long)NROWS * KIN) return;
  int c = (int)(idx % KIN);
  int row = (int)(idx / KIN);
  int n = row >> 12, l = row & 4095;
  float v;
  if (c < CONDD) {
    v = u3[(long)(n * 256 + (l >> 4)) * 512 + c];
  } else if (c < CONDD + EMB) {
    int e = xi[(long)row * 3 + 0];
    v = cemb[e * EMB + (c - CONDD)];
  } else if (c < CONDD + 2 * EMB) {
    int e = xi[(long)row * 3 + 1];
    v = femb[e * EMB + (c - CONDD - EMB)];
  } else {
    int e = xi[(long)row * 3 + 2];
    v = cemb[e * EMB + (c - CONDD - 2 * EMB)];
  }
  dst[idx] = (bhalf)v;
}

// ---------------------------------------------------------------------------
// Weight conversion kernels
// ---------------------------------------------------------------------------
__global__ void k_cvt_bf16(const float* __restrict__ s, bhalf* __restrict__ d,
                           long n) {
  long i = (long)blockIdx.x * blockDim.x + threadIdx.x;
  if (i < n) d[i] = (bhalf)s[i];
}

// I_W with the coarse/coarse_t autoregressive mask applied.
__global__ void k_cvt_iw(const float* __restrict__ s, bhalf* __restrict__ d) {
  long i = (long)blockIdx.x * blockDim.x + threadIdx.x;
  if (i >= (long)KIN * N3H) return;
  int k = (int)(i / N3H), j = (int)(i % N3H);
  bool masked = (k >= KIN - EMB) &&
                ((j < DHALF) || (j >= HDIM && j < HDIM + DHALF) ||
                 (j >= 2 * HDIM && j < 2 * HDIM + DHALF));
  d[i] = (bhalf)(masked ? 0.f : s[i]);
}

// ---------------------------------------------------------------------------
// Generic bf16 WMMA GEMM: C[M,N] = act(A[M,K](lda) * B[K,N](ldb) + bias)
// Block = 256 threads (8 waves), tile 64x64, K-step 64 (2 WMMA K-chunks).
// A tile staged via Tensor Data Mover (hardware-padded LDS rows);
// B tile staged with two 16B coalesced loads + LDS transpose per thread.
// ---------------------------------------------------------------------------
#define BM 64
#define BN 64
#define BK 64
#define APAD 72   // padded LDS row stride (matches TDM pad 32+4 DWORDs)

__launch_bounds__(256)
__global__ void k_gemm_bf16(const bhalf* __restrict__ A, int lda,
                            const bhalf* __restrict__ B, int ldb,
                            const float* __restrict__ bias,
                            void* __restrict__ Cout, int ldc,
                            int M, int N, int K, int relu, int obf16) {
  __shared__ bhalf As[BM * APAD];
  __shared__ bhalf BsT[BN * APAD];

  int gridN = N / BN;
  int bm = blockIdx.x / gridN;
  int bn = blockIdx.x % gridN;
  int tid = threadIdx.x, lane = tid & 31, w = tid >> 5;
  int sr  = w >> 1;            // subtile row 0..3
  int sc0 = (w & 1) * 2;       // subtile col base 0 or 2

  unsigned asLds = (unsigned)(size_t)(void*)As;   // LDS byte offset (addr[31:0])

  v8f acc0, acc1;
  #pragma unroll
  for (int i = 0; i < 8; i++) { acc0[i] = 0.f; acc1[i] = 0.f; }

  // B staging: K row skk (0..63), N chunk sn0 (0/16/32/48); running pointer.
  int skk = tid >> 2;
  int sn0 = (tid & 3) * 16;
  const bhalf* bptr = B + (size_t)skk * ldb + bn * BN + sn0;
  const bhalf* aptr = A + (size_t)(bm * BM) * lda;

  for (int k0 = 0; k0 < K; k0 += BK) {
    // Async TDM: A tile [64 rows x 64 bf16], row stride lda, padded in LDS.
    if (w == 0)
      tdm_load_2d_padded(asLds, aptr + k0, BK, BM, (unsigned long long)lda);
    // Stage B tile transposed: BsT[n][kk] = B[k0+kk][bn*64+n]
    v8bf bv0 = *(const v8bf*)bptr;
    v8bf bv1 = *(const v8bf*)(bptr + 8);
    #pragma unroll
    for (int j = 0; j < 8; j++) {
      BsT[(sn0 + j) * APAD + skk]     = bv0[j];
      BsT[(sn0 + 8 + j) * APAD + skk] = bv1[j];
    }
    bptr += (size_t)BK * ldb;
    if (w == 0) __builtin_amdgcn_s_wait_tensorcnt(0);
    __syncthreads();

    int m = sr * 16 + (lane & 15);
    int aoff = (lane & 16) ? 8 : 0;
    int boff = (lane & 16) ? 8 : 0;
    #pragma unroll
    for (int kc = 0; kc < 2; kc++) {
      const bhalf* ap = &As[m * APAD + kc * 32 + aoff];
      v8bf alo = *(const v8bf*)ap;
      v8bf ahi = *(const v8bf*)(ap + 16);
      v16bf afrag = __builtin_shufflevector(alo, ahi, 0, 1, 2, 3, 4, 5, 6, 7,
                                            8, 9, 10, 11, 12, 13, 14, 15);
      #pragma unroll
      for (int s = 0; s < 2; s++) {
        int n = (sc0 + s) * 16 + (lane & 15);
        const bhalf* bp = &BsT[n * APAD + kc * 32 + boff];
        v8bf blo = *(const v8bf*)bp;
        v8bf bhi = *(const v8bf*)(bp + 16);
        v16bf bfrag = __builtin_shufflevector(blo, bhi, 0, 1, 2, 3, 4, 5, 6, 7,
                                              8, 9, 10, 11, 12, 13, 14, 15);
        if (s == 0) acc0 = wmma_bf16(afrag, bfrag, acc0);
        else        acc1 = wmma_bf16(afrag, bfrag, acc1);
      }
    }
    __syncthreads();
  }

  // Epilogue: C 16x16 layout -> lane 0-15: M=v, N=lane; lane 16-31: M=8+v
  int mrow = bm * BM + sr * 16 + ((lane & 16) ? 8 : 0);
  #pragma unroll
  for (int s = 0; s < 2; s++) {
    v8f acc = s ? acc1 : acc0;
    int n = bn * BN + (sc0 + s) * 16 + (lane & 15);
    float bv = bias[n];
    if (obf16) {
      bhalf* C = (bhalf*)Cout;
      #pragma unroll
      for (int v = 0; v < 8; v++) {
        float val = acc[v] + bv;
        if (relu && val < 0.f) val = 0.f;
        C[(long)(mrow + v) * ldc + n] = (bhalf)val;
      }
    } else {
      float* C = (float*)Cout;
      #pragma unroll
      for (int v = 0; v < 8; v++) {
        float val = acc[v] + bv;
        if (relu && val < 0.f) val = 0.f;
        C[(long)(mrow + v) * ldc + n] = val;
      }
    }
  }
}

// ---------------------------------------------------------------------------
// Persistent cooperative GRU scan.
// 64 WGs x 256 threads. WG g owns hidden cols [16g,16g+16).
// Waves 0..5: gate (w%3), K-half (w/3); WMMA B-fragments for R_w held in regs.
// Wave 6: pointwise GRU update for 2 batches x 16 cols.
// Grid sync: monotonically increasing atomic counter (memset per launch).
// ---------------------------------------------------------------------------
#define HSTR 1032   // padded ldsH row stride (elements)

__launch_bounds__(256)
__global__ void k_scan(const float* __restrict__ Rw,   // [1024,3072]
                       const float* __restrict__ Rb,   // [3072]
                       const float* __restrict__ ix,   // [8192,3072]
                       float* __restrict__ hbuf,       // [2][2][1024]
                       bhalf* __restrict__ hs,         // [8192,1024]
                       unsigned* __restrict__ cnt) {
  __shared__ bhalf ldsH[16 * HSTR];          // A matrix, rows 2..15 stay zero
  __shared__ float gbufP[6][2][16];

  int tid = threadIdx.x, lane = tid & 31, w = tid >> 5;
  int j0 = blockIdx.x * 16;

  for (int i = tid; i < 16 * HSTR; i += 256) ldsH[i] = (bhalf)0.f;

  // Preload B fragments (R_w slice, bf16) into registers.
  v16bf bfrag[16];
  if (w < 6) {
    int g = w % 3, half = w / 3;
    int col = g * HDIM + j0 + (lane & 15);
    int kh  = half * 512 + ((lane & 16) ? 16 : 0);
    #pragma unroll
    for (int kc = 0; kc < 16; kc++) {
      #pragma unroll
      for (int e = 0; e < 16; e++)
        bfrag[kc][e] = (bhalf)Rw[(long)(kh + kc * 32 + e) * N3H + col];
    }
  }
  // Hoist loop-invariant recurrent gate biases (update wave only).
  float rbu = 0.f, rbr = 0.f, rbe = 0.f;
  if (w == 6) {
    int col = j0 + (lane & 15);
    rbu = Rb[col]; rbr = Rb[HDIM + col]; rbe = Rb[2 * HDIM + col];
  }
  __syncthreads();

  for (int t = 0; t < LSEQ; t++) {
    int p = t & 1;
    const float* hin = hbuf + p * (NBATCH * HDIM);
    float* hout = hbuf + (1 - p) * (NBATCH * HDIM);

    // Stage h (2x1024 f32 -> bf16) into LDS rows 0,1: 8 contiguous elements
    // per thread, one 16B ds_store_b128 each.
    {
      int base = tid * 8;
      const float* src = hin + base;
      v8bf hv;
      #pragma unroll
      for (int j = 0; j < 8; j++) hv[j] = (bhalf)src[j];
      *(v8bf*)&ldsH[(base >> 10) * HSTR + (base & 1023)] = hv;
    }
    __syncthreads();

    if (w < 6) {
      int half = w / 3;
      int m = lane & 15;
      int aoff = (lane & 16) ? 16 : 0;
      const bhalf* hp = &ldsH[m * HSTR + half * 512 + aoff];
      v8f acc;
      #pragma unroll
      for (int i = 0; i < 8; i++) acc[i] = 0.f;
      #pragma unroll
      for (int kc = 0; kc < 16; kc++) {
        const bhalf* q = hp + kc * 32;
        v8bf lo = *(const v8bf*)q;
        v8bf hi = *(const v8bf*)(q + 8);
        v16bf afrag = __builtin_shufflevector(lo, hi, 0, 1, 2, 3, 4, 5, 6, 7,
                                              8, 9, 10, 11, 12, 13, 14, 15);
        acc = wmma_bf16(afrag, bfrag[kc], acc);
      }
      if (lane < 16) {           // rows 0,1 of C = batch 0,1
        gbufP[w][0][lane] = acc[0];
        gbufP[w][1][lane] = acc[1];
      }
    }
    __syncthreads();

    if (w == 6) {
      int b = lane >> 4, j = lane & 15, col = j0 + j;
      long row = (long)b * LSEQ + t;
      const float* ixr = ix + row * N3H;
      float u1 = gbufP[0][b][j] + gbufP[3][b][j] + rbu;
      float r1 = gbufP[1][b][j] + gbufP[4][b][j] + rbr;
      float e1 = gbufP[2][b][j] + gbufP[5][b][j] + rbe;
      float ut = sigmoidf_(u1 + ixr[col]);
      float rt = sigmoidf_(r1 + ixr[HDIM + col]);
      float et = tanhf(rt * e1 + ixr[2 * HDIM + col]);
      float hx = hin[b * HDIM + col];
      float ht = ut * hx + (1.f - ut) * et;
      hout[b * HDIM + col] = ht;
      hs[row * HDIM + col] = (bhalf)ht;
    }

    // ---- grid-wide phase barrier ----
    __syncthreads();
    if (tid == 0) {
      __threadfence();
      atomicAdd(cnt, 1u);
      unsigned target = (unsigned)(t + 1) * SCAN_NWG;
      volatile unsigned* vc = cnt;
      while (*vc < target) __builtin_amdgcn_s_sleep(1);
    }
    __syncthreads();
    __threadfence();   // all waves: don't read stale h across the grid sync
  }
}

// ---------------------------------------------------------------------------
// log_softmax over 256 + interleave into [N,L,256,2]
// ---------------------------------------------------------------------------
__global__ void k_logsoftmax(const float* __restrict__ clog,
                             const float* __restrict__ flog,
                             float* __restrict__ out) {
  __shared__ float red[256];
  int b = blockIdx.x;
  int which = b & 1, row = b >> 1;
  const float* src = (which == 0 ? clog : flog) + (long)row * 256;
  int t = threadIdx.x;
  float v = src[t];
  red[t] = v; __syncthreads();
  for (int st = 128; st > 0; st >>= 1) {
    if (t < st) red[t] = fmaxf(red[t], red[t + st]);
    __syncthreads();
  }
  float mx = red[0]; __syncthreads();
  red[t] = __expf(v - mx); __syncthreads();
  for (int st = 128; st > 0; st >>= 1) {
    if (t < st) red[t] += red[t + st];
    __syncthreads();
  }
  float lse = logf(red[0]);
  out[((long)row * 256 + t) * 2 + which] = v - mx - lse;
}

// ---------------------------------------------------------------------------
// Host launch
// ---------------------------------------------------------------------------
static inline int ceil_div(long a, int b) { return (int)((a + b - 1) / b); }

extern "C" void kernel_launch(void* const* d_in, const int* in_sizes, int n_in,
                              void* d_out, int out_size, void* d_ws,
                              size_t ws_size, hipStream_t stream) {
  const int*   xi        = (const int*)  d_in[0];
  const float* mel       = (const float*)d_in[1];
  const float* conv_in_w = (const float*)d_in[2];
  const float* dconv1_w  = (const float*)d_in[3];
  const float* dconv2_w  = (const float*)d_in[4];
  const float* up1_w     = (const float*)d_in[5];
  const float* up2_w     = (const float*)d_in[6];
  const float* up3_w     = (const float*)d_in[7];
  const float* I_W       = (const float*)d_in[8];
  const float* I_b       = (const float*)d_in[9];
  const float* R_w       = (const float*)d_in[10];
  const float* R_b       = (const float*)d_in[11];
  const float* O1_w = (const float*)d_in[12];
  const float* O1_b = (const float*)d_in[13];
  const float* O2_w = (const float*)d_in[14];
  const float* O2_b = (const float*)d_in[15];
  const float* O3_w = (const float*)d_in[16];
  const float* O3_b = (const float*)d_in[17];
  const float* O4_w = (const float*)d_in[18];
  const float* O4_b = (const float*)d_in[19];
  const float* cemb = (const float*)d_in[20];
  const float* femb = (const float*)d_in[21];
  const float* bn_s[6]; const float* bn_o[6];
  for (int i = 0; i < 6; i++) {
    bn_s[i] = (const float*)d_in[22 + 2 * i];
    bn_o[i] = (const float*)d_in[23 + 2 * i];
  }

  char* base = (char*)d_ws;
  size_t off = 0;
  auto alloc = [&](size_t bytes) -> char* {
    char* p = base + off;
    off = (off + bytes + 255) & ~(size_t)255;
    return p;
  };
  unsigned* cnt  = (unsigned*)alloc(4);
  float* hbuf    = (float*)alloc(2 * NBATCH * HDIM * 4);
  size_t zero_bytes = off;                       // counter + h state
  float* c0   = (float*)alloc(2 * 22 * 512 * 4);
  float* r1b  = (float*)alloc(2 * 20 * 512 * 4);
  float* a1   = (float*)alloc(2 * 20 * 512 * 4);
  float* r2b  = (float*)alloc(2 * 16 * 512 * 4);
  float* a2   = (float*)alloc(2 * 16 * 512 * 4);
  float* u1b  = (float*)alloc(2 * 32 * 512 * 4);
  float* u2b  = (float*)alloc(2 * 64 * 512 * 4);
  float* u3b  = (float*)alloc(2 * 256 * 512 * 4);
  bhalf* inb  = (bhalf*)alloc((size_t)NROWS * KIN * 2);
  bhalf* Wi   = (bhalf*)alloc((size_t)KIN * N3H * 2);
  float* ix   = (float*)alloc((size_t)NROWS * N3H * 4);
  bhalf* hs   = (bhalf*)alloc((size_t)NROWS * HDIM * 2);
  bhalf* O1c  = (bhalf*)alloc(512 * 512 * 2);
  bhalf* O2c  = (bhalf*)alloc(512 * 256 * 2);
  bhalf* O3c  = (bhalf*)alloc(512 * 512 * 2);
  bhalf* O4c  = (bhalf*)alloc(512 * 256 * 2);
  bhalf* t1c  = (bhalf*)alloc((size_t)NROWS * 512 * 2);
  bhalf* t1f  = (bhalf*)alloc((size_t)NROWS * 512 * 2);
  float* clog = (float*)alloc((size_t)NROWS * 256 * 4);
  float* flog = (float*)alloc((size_t)NROWS * 256 * 4);

  hipMemsetAsync(d_ws, 0, zero_bytes, stream);

  // ---- Upsample network ----
  k_conv_in<<<ceil_div(2 * 22 * 512, 256), 256, 0, stream>>>(mel, conv_in_w, c0);
  k_bn_relu<<<512, 256, 0, stream>>>(c0, bn_s[0], bn_o[0], 512, 2 * 22);
  k_dconv<<<ceil_div(2 * 20 * 512, 256), 256, 0, stream>>>(c0, dconv1_w, r1b, 22, 20, 2);
  k_bn_relu<<<512, 256, 0, stream>>>(r1b, bn_s[1], bn_o[1], 512, 2 * 20);
  k_resadd<<<ceil_div(2 * 20 * 512, 256), 256, 0, stream>>>(c0, r1b, a1, 22, 20, 1);
  k_dconv<<<ceil_div(2 * 16 * 512, 256), 256, 0, stream>>>(a1, dconv2_w, r2b, 20, 16, 4);
  k_bn_relu<<<512, 256, 0, stream>>>(r2b, bn_s[2], bn_o[2], 512, 2 * 16);
  k_resadd<<<ceil_div(2 * 16 * 512, 256), 256, 0, stream>>>(a1, r2b, a2, 20, 16, 2);
  k_upsample<<<ceil_div(2 * 32 * 512, 256), 256, 0, stream>>>(a2, up1_w, u1b, 16, 2);
  k_bn_relu<<<512, 256, 0, stream>>>(u1b, bn_s[3], bn_o[3], 512, 2 * 32);
  k_upsample<<<ceil_div(2 * 64 * 512, 256), 256, 0, stream>>>(u1b, up2_w, u2b, 32, 2);
  k_bn_relu<<<512, 256, 0, stream>>>(u2b, bn_s[4], bn_o[4], 512, 2 * 64);
  k_upsample<<<ceil_div(2 * 256 * 512, 256), 256, 0, stream>>>(u2b, up3_w, u3b, 64, 4);
  k_bn_relu<<<512, 256, 0, stream>>>(u3b, bn_s[5], bn_o[5], 512, 2 * 256);

  // ---- Build GRU inputs + masked input weights, big WMMA GEMM for ix ----
  k_build_inputs<<<ceil_div((long)NROWS * KIN, 256), 256, 0, stream>>>(
      u3b, xi, cemb, femb, inb);
  k_cvt_iw<<<ceil_div((long)KIN * N3H, 256), 256, 0, stream>>>(I_W, Wi);
  k_gemm_bf16<<<(NROWS / BM) * (N3H / BN), 256, 0, stream>>>(
      inb, KIN, Wi, N3H, I_b, ix, N3H, NROWS, N3H, KIN, 0, 0);

  // ---- Persistent cooperative GRU scan ----
  k_scan<<<SCAN_NWG, 256, 0, stream>>>(R_w, R_b, ix, hbuf, hs, cnt);

  // ---- Output heads ----
  k_cvt_bf16<<<ceil_div(512 * 512, 256), 256, 0, stream>>>(O1_w, O1c, 512 * 512);
  k_cvt_bf16<<<ceil_div(512 * 256, 256), 256, 0, stream>>>(O2_w, O2c, 512 * 256);
  k_cvt_bf16<<<ceil_div(512 * 512, 256), 256, 0, stream>>>(O3_w, O3c, 512 * 512);
  k_cvt_bf16<<<ceil_div(512 * 256, 256), 256, 0, stream>>>(O4_w, O4c, 512 * 256);

  k_gemm_bf16<<<(NROWS / BM) * (512 / BN), 256, 0, stream>>>(
      hs, HDIM, O1c, 512, O1_b, t1c, 512, NROWS, 512, 512, 1, 1);
  k_gemm_bf16<<<(NROWS / BM) * (256 / BN), 256, 0, stream>>>(
      t1c, 512, O2c, 256, O2_b, clog, 256, NROWS, 256, 512, 0, 0);
  k_gemm_bf16<<<(NROWS / BM) * (512 / BN), 256, 0, stream>>>(
      hs + DHALF, HDIM, O3c, 512, O3_b, t1f, 512, NROWS, 512, 512, 1, 1);
  k_gemm_bf16<<<(NROWS / BM) * (256 / BN), 256, 0, stream>>>(
      t1f, 512, O4c, 256, O4_b, flog, 256, NROWS, 256, 512, 0, 0);

  // ---- log_softmax + interleave ----
  k_logsoftmax<<<NROWS * 2, 256, 0, stream>>>(clog, flog, (float*)d_out);
}
